// FactorizedTransformer_13391708029645
// MI455X (gfx1250) — matrix operand order repaired and checked by reading, not compiled, not run on hardware
//
#include <hip/hip_runtime.h>
#include <hip/hip_bf16.h>
#include <math.h>
#include <stdint.h>

// =====================================================================
// FactorizedTransformer forward for MI455X (gfx1250, wave32, WMMA).
//
// Assumed d_in layout (setup_inputs dict order; params pytree-flattened
// in sorted-key order, uppercase before lowercase):
//   0:u 1:pos_x 2:pos_y 3:Win 4:Wo1 5:Wo2 6:Wv 7:fourier_B 8:gn_b 9:gn_g
//   10:kx_W 11:kx_g 12:ky_W 13:ky_g 14:ln_b 15:ln_g 16:pos_W 17:pos_b
//   18:tx.W1 19:tx.W2 20:tx.Win 21:tx.b1 22:tx.b2 23:tx.ln_b 24:tx.ln_g
//   25:ty.W1 26:ty.W2 27:ty.Win 28:ty.b1 29:ty.b2 30:ty.ln_b 31:ty.ln_g
//
// Workspace requirement: ~270 MB.
// =====================================================================

typedef __attribute__((ext_vector_type(16))) __bf16 v16bf;
typedef __attribute__((ext_vector_type(8)))  __bf16 v8bf;
typedef __attribute__((ext_vector_type(8)))  float  v8f;

// CDNA5 data-movement feature detection (host pass sees none -> fallback).
#if __has_builtin(__builtin_amdgcn_global_load_async_to_lds_b128) && \
    __has_builtin(__builtin_amdgcn_s_wait_asynccnt)
#define FT_ASYNC 1
#else
#define FT_ASYNC 0
#endif
#if __has_builtin(__builtin_amdgcn_tensor_load_to_lds) && \
    __has_builtin(__builtin_amdgcn_s_wait_tensorcnt)
#define FT_TDM 1
#else
#define FT_TDM 0
#endif

typedef __attribute__((ext_vector_type(4))) unsigned int u32x4;
typedef __attribute__((ext_vector_type(8))) int          i32x8;
typedef __attribute__((ext_vector_type(4))) int          i32x4;

#if FT_ASYNC
// The builtin's params are GCC-vector int4 pointers in as(1)/as(3)
// (clang prints as(1) as "__device__" in diagnostics).
typedef int ft_v4i __attribute__((vector_size(16)));
typedef __attribute__((address_space(1))) ft_v4i* ft_gv4p;
typedef __attribute__((address_space(3))) ft_v4i* ft_lv4p;
// async global -> LDS, 16B per lane, tracked by ASYNCcnt
__device__ __forceinline__ void ft_async_b128(const void* gsrc, void* ldst) {
  __builtin_amdgcn_global_load_async_to_lds_b128(
      (ft_gv4p)(unsigned long long)(uintptr_t)gsrc,
      (ft_lv4p)(unsigned int)(uintptr_t)ldst, 0, 0);
}
#endif

enum { EP_F32 = 0, EP_BF16 = 1, EP_GELU_F32 = 2, EP_GELU_BF16 = 3,
       EP_RESID = 4, EP_POSADD = 5 };

__device__ __forceinline__ float gelu_f(float x) {
  return 0.5f * x * (1.0f + erff(x * 0.70710678118654752f));
}

__device__ __forceinline__ v8f zero8() {
  v8f z = {0.f, 0.f, 0.f, 0.f, 0.f, 0.f, 0.f, 0.f};
  return z;
}

// ---------------------------------------------------------------------
// Generic GEMM: C(M,N) = A(M,K) @ W(N,K)^T, bf16 operands, f32 acc.
// BM=128, BN=64, BK=32. 8 waves, each wave a 32x32 subtile (2x2 WMMA).
// Tile staging uses GLOBAL_LOAD_ASYNC_TO_LDS when available.
// ---------------------------------------------------------------------
__global__ __launch_bounds__(256) void ft_gemm_nt(
    const __bf16* __restrict__ A, const __bf16* __restrict__ W,
    const float* __restrict__ bias, void* __restrict__ Cout,
    int M, int N, int K, int epi)
{
  __shared__ __attribute__((aligned(32))) __bf16 As[128 * 32];
  __shared__ __attribute__((aligned(32))) __bf16 Bs[64 * 32];

  const int tid = threadIdx.x;
  const int m0 = blockIdx.y * 128;
  const int n0 = blockIdx.x * 64;

  const int wv = tid >> 5, lane = tid & 31;
  const int wm = wv & 3, wn = wv >> 2;          // 4x2 wave grid
  const int l16 = lane & 15, hl = lane >> 4;

  v8f acc[2][2];
  acc[0][0] = zero8(); acc[0][1] = zero8();
  acc[1][0] = zero8(); acc[1][1] = zero8();

  for (int k0 = 0; k0 < K; k0 += 32) {
    {   // A tile: 128x32, every thread stages a half-row (16 bf16)
      int r = tid >> 1, kh = (tid & 1) * 16;
      const __bf16* src = A + (size_t)(m0 + r) * K + k0 + kh;
      if (k0 + 32 < K) __builtin_prefetch(src + 32, 0, 3);
#if FT_ASYNC
      ft_async_b128(src,     As + r * 32 + kh);
      ft_async_b128(src + 8, As + r * 32 + kh + 8);
#else
      *(v8bf*)(As + r * 32 + kh)     = *(const v8bf*)(src);
      *(v8bf*)(As + r * 32 + kh + 8) = *(const v8bf*)(src + 8);
#endif
    }
    if (tid < 128) {  // B tile: 64x32 from N-major weight (coalesced)
      int n = tid >> 1, kh = (tid & 1) * 16;
      const __bf16* src = W + (size_t)(n0 + n) * K + k0 + kh;
#if FT_ASYNC
      ft_async_b128(src,     Bs + n * 32 + kh);
      ft_async_b128(src + 8, Bs + n * 32 + kh + 8);
#else
      *(v8bf*)(Bs + n * 32 + kh)     = *(const v8bf*)(src);
      *(v8bf*)(Bs + n * 32 + kh + 8) = *(const v8bf*)(src + 8);
#endif
    }
#if FT_ASYNC
    __builtin_amdgcn_s_wait_asynccnt(0);
#endif
    __syncthreads();
#pragma unroll
    for (int mt = 0; mt < 2; ++mt) {
      union { v16bf v; v8bf h[2]; } af;
      const __bf16* ap = As + (wm * 32 + mt * 16 + l16) * 32;
      af.h[0] = *(const v8bf*)(ap + hl * 16);       // K = hl*16 .. +7
      af.h[1] = *(const v8bf*)(ap + hl * 16 + 8);   // K = hl*16+8 .. +15
#pragma unroll
      for (int nt = 0; nt < 2; ++nt) {
        v16bf bf = *(const v16bf*)(Bs + (wn * 32 + nt * 16 + l16) * 32 + hl * 16);
        acc[mt][nt] = __builtin_amdgcn_wmma_f32_16x16x32_bf16(
            false, af.v, false, bf, (short)0, acc[mt][nt], false, false);
      }
    }
    __syncthreads();
  }

  float*  Cf = (float*)Cout;
  __bf16* Cb = (__bf16*)Cout;
#pragma unroll
  for (int mt = 0; mt < 2; ++mt)
#pragma unroll
    for (int nt = 0; nt < 2; ++nt) {
      int ncol = n0 + wn * 32 + nt * 16 + l16;
      float bv = bias ? bias[ncol] : 0.0f;
#pragma unroll
      for (int e = 0; e < 8; ++e) {
        int mrow = m0 + wm * 32 + mt * 16 + hl * 8 + e;
        float x = acc[mt][nt][e] + bv;
        size_t idx = (size_t)mrow * N + ncol;
        if (epi == EP_GELU_F32 || epi == EP_GELU_BF16) x = gelu_f(x);
        if (epi == EP_F32 || epi == EP_GELU_F32) {
          Cf[idx] = x;
        } else if (epi == EP_BF16 || epi == EP_GELU_BF16) {
          Cb[idx] = (__bf16)x;
        } else if (epi == EP_RESID) {
          Cf[idx] = Cf[idx] + x;
        } else { // EP_POSADD: broadcast-add across the 4 batches of u
          for (int bb = 0; bb < 4; ++bb) {
            size_t o = (size_t)bb * (size_t)M * N + idx;
            Cf[o] = Cf[o] + x;
          }
        }
      }
    }
}

// ---------------------------------------------------------------------
// Factorized-attention einsum: per (b,h): C(128, 8192) = A(128,128) @ B.
// A (kernel matrix) loaded LDS-resident via the Tensor Data Mover when
// available (32KB contiguous tile, one TDM descriptor issued by wave 0,
// completion via TENSORcnt); B gathered with runtime strides.
// n decomposes as (outer, c<64); BN tile = 64 => one outer per block.
// ---------------------------------------------------------------------
__global__ __launch_bounds__(256) void ft_einsum(
    const __bf16* __restrict__ Kmat, const __bf16* __restrict__ Bsrc,
    __bf16* __restrict__ Out,
    long long offB_b, long long offB_h, long long sK, long long sOutB,
    long long offO_b, long long offO_h, long long sOM, long long sOutO)
{
  __shared__ __attribute__((aligned(32))) __bf16 Ak[128 * 128];
  __shared__ __attribute__((aligned(32))) __bf16 Bs[64 * 32];

  const int tid = threadIdx.x;
  const int outer = blockIdx.x;     // 0..127
  const int bh = blockIdx.y;        // 0..31
  const int b = bh >> 3, h = bh & 7;

  const __bf16* Abh = Kmat + (size_t)bh * 16384;
#if FT_TDM
  if (tid < 32) {   // one wave issues the TDM descriptor for the 32KB tile
    unsigned long long ga = (unsigned long long)(uintptr_t)Abh;
    unsigned int ldsoff = (unsigned int)(uintptr_t)Ak;   // LDS byte offset
    // D# group 0: count=1, lds_addr, global_addr[56:0], type=2
    u32x4 g0 = { 1u, ldsoff, (unsigned int)ga,
                 (unsigned int)((ga >> 32) & 0x01FFFFFFu) | (2u << 30) };
    // D# group 1: data_size=2B; tensor_dim0=16384, tensor_dim1=1,
    // tile_dim0=16384, tile_dim1=1, stride0=16384 (contiguous run)
    i32x8 g1 = { (int)0x00010000,
                 (int)(16384u << 16),            // tensor_dim0[15:0] @63:48
                 (int)(0u | (1u << 16)),         // dim0 hi | tensor_dim1 lo
                 (int)(0u | (16384u << 16)),     // dim1 hi | tile_dim0
                 (int)(1u),                      // tile_dim1=1, tile_dim2=0
                 (int)16384, 0, 0 };             // tensor_dim0_stride
    i32x4 gz = {0, 0, 0, 0};
#if __has_include(<hip/amd_detail/amd_gfx1250_TDM.h>)
    i32x8 gz8 = {0, 0, 0, 0, 0, 0, 0, 0};
    __builtin_amdgcn_tensor_load_to_lds(g0, g1, gz, gz, gz8, 0);
#else
    __builtin_amdgcn_tensor_load_to_lds(g0, g1, gz, gz, 0);
#endif
    __builtin_amdgcn_s_wait_tensorcnt(0);
  }
#elif FT_ASYNC
  for (int c = tid; c < 2048; c += 256)
    ft_async_b128(Abh + c * 8, Ak + c * 8);
  __builtin_amdgcn_s_wait_asynccnt(0);
#else
  for (int c = tid; c < 2048; c += 256)
    *(v8bf*)(Ak + c * 8) = *(const v8bf*)(Abh + c * 8);
#endif

  const long long baseB = offB_b * b + offB_h * h + sOutB * outer;

  const int wv = tid >> 5, lane = tid & 31;
  const int wm = wv & 3, wn = wv >> 2;
  const int l16 = lane & 15, hl = lane >> 4;

  v8f acc[2][2];
  acc[0][0] = zero8(); acc[0][1] = zero8();
  acc[1][0] = zero8(); acc[1][1] = zero8();

  for (int k0 = 0; k0 < 128; k0 += 32) {
    __syncthreads();   // Ak ready (first iter) / Bs consumed (later iters)
    {   // gather B(kk, c) -> transposed LDS tile Bs[c][kk]
      int kk = tid & 31, cb = (tid >> 5) * 8;
      const __bf16* src = Bsrc + baseB + (long long)(k0 + kk) * sK + cb;
      if (k0 + 32 < 128) __builtin_prefetch(src + 32 * sK, 0, 3);
      v8bf t = *(const v8bf*)src;
#pragma unroll
      for (int j = 0; j < 8; ++j) Bs[(cb + j) * 32 + kk] = t[j];
    }
    __syncthreads();
#pragma unroll
    for (int mt = 0; mt < 2; ++mt) {
      union { v16bf v; v8bf h[2]; } af;
      const __bf16* ap = Ak + (wm * 32 + mt * 16 + l16) * 128 + k0;
      af.h[0] = *(const v8bf*)(ap + hl * 16);
      af.h[1] = *(const v8bf*)(ap + hl * 16 + 8);
#pragma unroll
      for (int nt = 0; nt < 2; ++nt) {
        v16bf bf = *(const v16bf*)(Bs + (wn * 32 + nt * 16 + l16) * 32 + hl * 16);
        acc[mt][nt] = __builtin_amdgcn_wmma_f32_16x16x32_bf16(
            false, af.v, false, bf, (short)0, acc[mt][nt], false, false);
      }
    }
  }

  const long long baseO = offO_b * b + offO_h * h + sOutO * outer;
#pragma unroll
  for (int mt = 0; mt < 2; ++mt)
#pragma unroll
    for (int nt = 0; nt < 2; ++nt) {
      int ncol = wn * 32 + nt * 16 + l16;
#pragma unroll
      for (int e = 0; e < 8; ++e) {
        int mrow = wm * 32 + mt * 16 + hl * 8 + e;
        Out[baseO + (long long)mrow * sOM + ncol] = (__bf16)acc[mt][nt][e];
      }
    }
}

// ---------------------------------------------------------------------
// LayerNorm over last dim (256). One block per row.
// ---------------------------------------------------------------------
__global__ __launch_bounds__(256) void ft_ln(
    const float* __restrict__ X, const float* __restrict__ g,
    const float* __restrict__ bb, __bf16* __restrict__ outb,
    float* __restrict__ outf, int to_bf)
{
  __shared__ float red[256];
  const int row = blockIdx.x, tid = threadIdx.x;
  float x = X[(size_t)row * 256 + tid];
  red[tid] = x; __syncthreads();
  for (int s = 128; s > 0; s >>= 1) {
    if (tid < s) red[tid] += red[tid + s];
    __syncthreads();
  }
  float mean = red[0] * (1.0f / 256.0f);
  __syncthreads();
  float d = x - mean;
  red[tid] = d * d; __syncthreads();
  for (int s = 128; s > 0; s >>= 1) {
    if (tid < s) red[tid] += red[tid + s];
    __syncthreads();
  }
  float var = red[0] * (1.0f / 256.0f);
  float y = d * rsqrtf(var + 1e-5f) * g[tid] + bb[tid];
  if (to_bf) outb[(size_t)row * 256 + tid] = (__bf16)y;
  else       outf[(size_t)row * 256 + tid] = y;
}

// mean over one spatial axis of un (b, 128, 128, 256) -> (4*128, 256)
__global__ void ft_mean(const __bf16* __restrict__ un, float* __restrict__ out,
                        long long rowStride, long long sumStride)
{
  int t = blockIdx.x * 256 + threadIdx.x;
  if (t >= 131072) return;
  int c = t & 255, r = t >> 8;
  int b = r >> 7, xy = r & 127;
  long long base = (long long)b * 4194304 + (long long)xy * rowStride + c;
  float s = 0.f;
  for (int k = 0; k < 128; ++k) s += (float)un[base + (long long)k * sumStride];
  out[t] = s * (1.0f / 128.0f);
}

// naive small GEMM: C(rows,N) = X(rows,K) @ W(K,N) [+bias][gelu], f32
__global__ void ft_mm_small(const float* __restrict__ X, const float* __restrict__ W,
                            const float* __restrict__ bias, float* __restrict__ C,
                            int rows, int K, int N, int do_gelu)
{
  int t = blockIdx.x * 256 + threadIdx.x;
  if (t >= rows * N) return;
  int n = t % N, r = t / N;
  const float* xp = X + (size_t)r * K;
  float s = bias ? bias[n] : 0.f;
  for (int k = 0; k < K; ++k) s += xp[k] * W[(size_t)k * N + n];
  C[t] = do_gelu ? gelu_f(s) : s;
}

// fourier positional features -> bf16 (16384, 256)
__global__ void ft_posfeat(const float* __restrict__ px, const float* __restrict__ py,
                           const float* __restrict__ B, __bf16* __restrict__ feat)
{
  int t = blockIdx.x * 256 + threadIdx.x;
  if (t >= 4194304) return;
  int c = t & 255, xy = t >> 8;
  int x = xy >> 7, y = xy & 127;
  int j = c & 127;
  float proj = 6.283185307179586f * (px[x] * B[j] + py[y] * B[128 + j]);
  feat[t] = (__bf16)((c < 128) ? sinf(proj) : cosf(proj));
}

// rope: qk (4,128,3072) -> q,k (4,8,128,192) f32, pos 128
__global__ void ft_rope(const float* __restrict__ qk, const float* __restrict__ pos,
                        float* __restrict__ qo, float* __restrict__ ko)
{
  int t = blockIdx.x * 256 + threadIdx.x;
  if (t >= 786432) return;
  int d = t % 192;
  int i = (t / 192) % 128;
  int hh = (t / (192 * 128)) % 8;
  int b = t / (192 * 128 * 8);
  int j = d % 96;
  float inv = powf(10000.0f, -(2.0f * j) / 192.0f);
  float f = pos[i] * 64.0f * inv;   // pos / MIN_FREQ, MIN_FREQ = 1/64
  float cs = cosf(f), sn = sinf(f);
  size_t rowq = (size_t)(b * 128 + i) * 3072 + hh * 192;
  float qv = qk[rowq + d];
  float kv = qk[rowq + 1536 + d];
  int dob = (d < 96) ? d + 96 : d - 96;
  float qr = qk[rowq + dob];
  float kr = qk[rowq + 1536 + dob];
  if (d < 96) { qr = -qr; kr = -kr; }
  size_t oidx = (size_t)((b * 8 + hh) * 128 + i) * 192 + d;
  qo[oidx] = qv * cs + qr * sn;
  ko[oidx] = kv * cs + kr * sn;
}

// kernel matrix: K[bh,i,j] = q_i . k_j + (i==j)*gamma, output bf16
__global__ void ft_kmat(const float* __restrict__ q, const float* __restrict__ k,
                        const float* __restrict__ gptr, __bf16* __restrict__ Kout)
{
  int t = blockIdx.x * 256 + threadIdx.x;
  if (t >= 524288) return;
  int j = t & 127, i = (t >> 7) & 127, bh = t >> 14;
  const float* qp = q + (size_t)(bh * 128 + i) * 192;
  const float* kp = k + (size_t)(bh * 128 + j) * 192;
  float s = 0.f;
  for (int d = 0; d < 192; ++d) s += qp[d] * kp[d];
  if (i == j) s += gptr[0];
  Kout[t] = (__bf16)s;
}

// group-norm stats per (b,h) over (i,l,c) of o2 (b,i,l,512) bf16
__global__ __launch_bounds__(256) void ft_gn_reduce(const __bf16* __restrict__ o2,
                                                    float* __restrict__ stats)
{
  int bh = blockIdx.x, tid = threadIdx.x;
  int b = bh >> 3, h = bh & 7;
  size_t base = (size_t)b * 8388608 + (size_t)h * 64;
  float s = 0.f, s2 = 0.f;
  for (int e = tid; e < 1048576; e += 256) {
    int c = e & 63, il = e >> 6;
    float v = (float)o2[base + (size_t)il * 512 + c];
    s += v; s2 += v * v;
  }
  __shared__ float r1[256], r2[256];
  r1[tid] = s; r2[tid] = s2; __syncthreads();
  for (int st = 128; st > 0; st >>= 1) {
    if (tid < st) { r1[tid] += r1[tid + st]; r2[tid] += r2[tid + st]; }
    __syncthreads();
  }
  if (tid == 0) {
    float m = r1[0] * (1.0f / 1048576.0f);
    stats[bh * 2]     = m;
    stats[bh * 2 + 1] = r2[0] * (1.0f / 1048576.0f) - m * m;
  }
}

__global__ void ft_gn_apply(const __bf16* __restrict__ o2, const float* __restrict__ stats,
                            const float* __restrict__ gng, const float* __restrict__ gnb,
                            __bf16* __restrict__ out)
{
  int t = blockIdx.x * 256 + threadIdx.x;
  if (t >= 33554432) return;
  int col = t & 511;
  int b = t >> 23;
  int h = col >> 6;
  float m = stats[(b * 8 + h) * 2];
  float v = stats[(b * 8 + h) * 2 + 1];
  float x = ((float)o2[t] - m) * rsqrtf(v + 1e-5f) * gng[col] + gnb[col];
  out[t] = (__bf16)x;
}

// transpose+convert weight: in f32 (K,N) -> out bf16 (N,K)
__global__ void ft_convT(const float* __restrict__ in, __bf16* __restrict__ out,
                         int K, int N)
{
  int t = blockIdx.x * 256 + threadIdx.x;
  if (t >= K * N) return;
  int k = t % K, n = t / K;
  out[t] = (__bf16)in[(size_t)k * N + n];
}

// =====================================================================
extern "C" void kernel_launch(void* const* d_in, const int* in_sizes, int n_in,
                              void* d_out, int out_size, void* d_ws, size_t ws_size,
                              hipStream_t stream)
{
  (void)in_sizes; (void)n_in; (void)out_size; (void)ws_size;

  const float* u_in      = (const float*)d_in[0];
  const float* pos_x     = (const float*)d_in[1];
  const float* pos_y     = (const float*)d_in[2];
  const float* Win       = (const float*)d_in[3];
  const float* Wo1       = (const float*)d_in[4];
  const float* Wo2       = (const float*)d_in[5];
  const float* Wv        = (const float*)d_in[6];
  const float* fourier_B = (const float*)d_in[7];
  const float* gn_b      = (const float*)d_in[8];
  const float* gn_g      = (const float*)d_in[9];
  const float* kx_W      = (const float*)d_in[10];
  const float* kx_g      = (const float*)d_in[11];
  const float* ky_W      = (const float*)d_in[12];
  const float* ky_g      = (const float*)d_in[13];
  const float* ln_b      = (const float*)d_in[14];
  const float* ln_g      = (const float*)d_in[15];
  const float* pos_W     = (const float*)d_in[16];
  const float* pos_b     = (const float*)d_in[17];
  const float* txW1      = (const float*)d_in[18];
  const float* txW2      = (const float*)d_in[19];
  const float* txWin     = (const float*)d_in[20];
  const float* txb1      = (const float*)d_in[21];
  const float* txb2      = (const float*)d_in[22];
  const float* txlnb     = (const float*)d_in[23];
  const float* txlng     = (const float*)d_in[24];
  const float* tyW1      = (const float*)d_in[25];
  const float* tyW2      = (const float*)d_in[26];
  const float* tyWin     = (const float*)d_in[27];
  const float* tyb1      = (const float*)d_in[28];
  const float* tyb2      = (const float*)d_in[29];
  const float* tylnb     = (const float*)d_in[30];
  const float* tylng     = (const float*)d_in[31];

  float* u = (float*)d_out;  // running residual stream, f32 (4,128,128,256)

  char* wbase = (char*)d_ws; size_t off = 0;
  auto alloc = [&](size_t bytes) -> void* {
    void* p = wbase + off;
    off += (bytes + 255) & ~(size_t)255;
    return p;
  };

  __bf16* un    = (__bf16*)alloc(33554432);   // (65536,256) bf16
  __bf16* vb    = (__bf16*)alloc(67108864);   // (65536,512) bf16, reused as on
  __bf16* o1    = (__bf16*)alloc(67108864);   // (4,8,128,128,64) bf16
  __bf16* o2    = (__bf16*)alloc(67108864);   // (4,128,128,512) bf16
  __bf16* feat  = (__bf16*)alloc(8388608);    // (16384,256) bf16
  __bf16* posWT = (__bf16*)alloc(131072);
  __bf16* WvT   = (__bf16*)alloc(262144);
  __bf16* Wo1T  = (__bf16*)alloc(262144);
  __bf16* Wo2T  = (__bf16*)alloc(131072);
  float*  rmean = (float*)alloc(524288);      // (512,256)
  float*  cmean = (float*)alloc(524288);
  float*  tA    = (float*)alloc(524288);
  float*  tB    = (float*)alloc(524288);
  float*  uxb   = (float*)alloc(524288);
  float*  uyb   = (float*)alloc(524288);
  float*  qk    = (float*)alloc(6291456);     // (512,3072)
  float*  qxr   = (float*)alloc(3145728);     // (4,8,128,192)
  float*  kxr   = (float*)alloc(3145728);
  float*  qyr   = (float*)alloc(3145728);
  float*  kyr   = (float*)alloc(3145728);
  __bf16* Kx    = (__bf16*)alloc(1048576);    // (32,128,128) bf16
  __bf16* Ky    = (__bf16*)alloc(1048576);
  float*  stats = (float*)alloc(256);
  __bf16* g1    = un;  // reuse (un dead after means)
  __bf16* on    = vb;  // reuse (v dead after einsum1)

  (void)hipMemcpyAsync(u, u_in, (size_t)16777216 * 4, hipMemcpyDeviceToDevice, stream);

  for (int L = 0; L < 4; ++L) {
    // --- bf16 N-major weight copies for WMMA GEMMs ---
    ft_convT<<<256, 256, 0, stream>>>(pos_W + (size_t)L * 65536,  posWT, 256, 256);
    ft_convT<<<512, 256, 0, stream>>>(Wv    + (size_t)L * 131072, WvT,   256, 512);
    ft_convT<<<512, 256, 0, stream>>>(Wo1   + (size_t)L * 131072, Wo1T,  512, 256);
    ft_convT<<<256, 256, 0, stream>>>(Wo2   + (size_t)L * 65536,  Wo2T,  256, 256);

    // --- fourier pos-embed: u += feat @ pos_W + pos_b (broadcast batch) ---
    ft_posfeat<<<16384, 256, 0, stream>>>(pos_x, pos_y, fourier_B + (size_t)L * 256, feat);
    ft_gemm_nt<<<dim3(4, 128), 256, 0, stream>>>(feat, posWT, pos_b + (size_t)L * 256,
                                                 u, 16384, 256, 256, EP_POSADD);

    // --- layernorm -> un (bf16) ---
    ft_ln<<<65536, 256, 0, stream>>>(u, ln_g + (size_t)L * 256, ln_b + (size_t)L * 256,
                                     un, nullptr, 1);

    // --- v = un @ Wv (bf16 out) ---
    ft_gemm_nt<<<dim3(8, 512), 256, 0, stream>>>(un, WvT, nullptr, vb,
                                                 65536, 512, 256, EP_BF16);

    // --- axis means of un (matmul/mean commute => skip full ui) ---
    ft_mean<<<512, 256, 0, stream>>>(un, rmean, 32768LL, 256LL);   // mean over y
    ft_mean<<<512, 256, 0, stream>>>(un, cmean, 256LL, 32768LL);   // mean over x

    // --- x pooling chain + kernel matrix ---
    ft_mm_small<<<512, 256, 0, stream>>>(rmean, Win + (size_t)L * 65536, nullptr, tA, 512, 256, 256, 0);
    ft_mm_small<<<512, 256, 0, stream>>>(tA, txWin + (size_t)L * 65536, nullptr, tB, 512, 256, 256, 0);
    ft_ln<<<512, 256, 0, stream>>>(tB, txlng + (size_t)L * 256, txlnb + (size_t)L * 256, nullptr, tA, 0);
    ft_mm_small<<<512, 256, 0, stream>>>(tA, txW1 + (size_t)L * 65536, txb1 + (size_t)L * 256, tB, 512, 256, 256, 1);
    ft_mm_small<<<512, 256, 0, stream>>>(tB, txW2 + (size_t)L * 65536, txb2 + (size_t)L * 256, uxb, 512, 256, 256, 0);
    ft_mm_small<<<6144, 256, 0, stream>>>(uxb, kx_W + (size_t)L * 786432, nullptr, qk, 512, 256, 3072, 0);
    ft_rope<<<3072, 256, 0, stream>>>(qk, pos_x, qxr, kxr);
    ft_kmat<<<2048, 256, 0, stream>>>(qxr, kxr, kx_g + L, Kx);

    // --- y pooling chain + kernel matrix ---
    ft_mm_small<<<512, 256, 0, stream>>>(cmean, Win + (size_t)L * 65536, nullptr, tA, 512, 256, 256, 0);
    ft_mm_small<<<512, 256, 0, stream>>>(tA, tyWin + (size_t)L * 65536, nullptr, tB, 512, 256, 256, 0);
    ft_ln<<<512, 256, 0, stream>>>(tB, tylng + (size_t)L * 256, tylnb + (size_t)L * 256, nullptr, tA, 0);
    ft_mm_small<<<512, 256, 0, stream>>>(tA, tyW1 + (size_t)L * 65536, tyb1 + (size_t)L * 256, tB, 512, 256, 256, 1);
    ft_mm_small<<<512, 256, 0, stream>>>(tB, tyW2 + (size_t)L * 65536, tyb2 + (size_t)L * 256, uyb, 512, 256, 256, 0);
    ft_mm_small<<<6144, 256, 0, stream>>>(uyb, ky_W + (size_t)L * 786432, nullptr, qk, 512, 256, 3072, 0);
    ft_rope<<<3072, 256, 0, stream>>>(qk, pos_y, qyr, kyr);
    ft_kmat<<<2048, 256, 0, stream>>>(qyr, kyr, ky_g + L, Ky);

    // --- einsum 1: o1[b,h,i,l,c] = sum_j Kx[i,j] * v[b,j,l,h*64+c] ---
    ft_einsum<<<dim3(128, 32), 256, 0, stream>>>(Kx, vb, o1,
        /*offB_b*/ 8388608LL, /*offB_h*/ 64LL, /*sK*/ 65536LL, /*sOutB*/ 512LL,
        /*offO_b*/ 8388608LL, /*offO_h*/ 1048576LL, /*sOM*/ 8192LL, /*sOutO*/ 64LL);

    // --- einsum 2: o2[b,i,l,h*64+c] = sum_m Ky[l,m] * o1[b,h,i,m,c] ---
    ft_einsum<<<dim3(128, 32), 256, 0, stream>>>(Ky, o1, o2,
        /*offB_b*/ 8388608LL, /*offB_h*/ 1048576LL, /*sK*/ 64LL, /*sOutB*/ 8192LL,
        /*offO_b*/ 8388608LL, /*offO_h*/ 64LL, /*sOM*/ 512LL, /*sOutO*/ 65536LL);

    // --- group norm over (i,l,c) per (b,h) ---
    ft_gn_reduce<<<32, 256, 0, stream>>>(o2, stats);
    ft_gn_apply<<<131072, 256, 0, stream>>>(o2, stats, gn_g + (size_t)L * 512,
                                            gn_b + (size_t)L * 512, on);

    // --- g1 = gelu(on @ Wo1) (bf16) ---
    ft_gemm_nt<<<dim3(4, 512), 256, 0, stream>>>(on, Wo1T, nullptr, g1,
                                                 65536, 256, 512, EP_GELU_BF16);

    // --- u += g1 @ Wo2 (residual) ---
    ft_gemm_nt<<<dim3(4, 512), 256, 0, stream>>>(g1, Wo2T, nullptr, u,
                                                 65536, 256, 256, EP_RESID);
  }
}
